// SelfAttention_3753801416865
// MI455X (gfx1250) — matrix operand order repaired and checked by reading, not compiled, not run on hardware
//
#include <hip/hip_runtime.h>

#define BQ 8
#define SQ 2048
#define DQ 768

typedef __attribute__((ext_vector_type(16))) _Float16 v16h;
typedef __attribute__((ext_vector_type(8)))  _Float16 v8h;
typedef __attribute__((ext_vector_type(4)))  _Float16 v4h;
typedef __attribute__((ext_vector_type(8)))  float    v8f;

// ---------------------------------------------------------------------------
// CDNA5 async global->LDS copy (ASYNCcnt-tracked DMA), inline asm.
// ---------------------------------------------------------------------------
__device__ __forceinline__ void async_copy_b128(unsigned ldsOff, const _Float16* g) {
    asm volatile("global_load_async_to_lds_b128 %0, %1, off"
                 :: "v"(ldsOff), "v"((unsigned long long)(uintptr_t)g)
                 : "memory");
}
__device__ __forceinline__ void wait_asynccnt0() {
    asm volatile("s_wait_asynccnt 0x0" ::: "memory");
}

// ---------------------------------------------------------------------------
// f32 -> f16 conversion (vectorized x4)
// ---------------------------------------------------------------------------
__global__ void __launch_bounds__(256) cvt_f32_to_f16(const float* __restrict__ src,
                                                      _Float16* __restrict__ dst,
                                                      int n4) {
    int i = blockIdx.x * 256 + threadIdx.x;
    if (i < n4) {
        float4 v = ((const float4*)src)[i];
        v4h p = { (_Float16)v.x, (_Float16)v.y, (_Float16)v.z, (_Float16)v.w };
        ((v4h*)dst)[i] = p;
    }
}

// ---------------------------------------------------------------------------
// One 32-wide K step: 2 A-fragments x 4 B-tiles = 8 WMMAs into a single
// accumulator web (both unrolled halves write the same acc variables).
// ---------------------------------------------------------------------------
__device__ __forceinline__ void mma_step(const _Float16* __restrict__ aSrc0,
                                         const _Float16* __restrict__ aSrc1,
                                         const _Float16* __restrict__ bl,
                                         v8f acc[2][4]) {
    // A fragments, contiguous K (matches B):
    //   lanes 0-15: K {0..15}; lanes 16-31: K {16..31}
    v8h a0lo = *(const v8h*)(aSrc0);
    v8h a0hi = *(const v8h*)(aSrc0 + 8);
    v8h a1lo = *(const v8h*)(aSrc1);
    v8h a1hi = *(const v8h*)(aSrc1 + 8);
    v16h a0 = __builtin_shufflevector(a0lo, a0hi, 0,1,2,3,4,5,6,7,
                                                  8,9,10,11,12,13,14,15);
    v16h a1 = __builtin_shufflevector(a1lo, a1hi, 0,1,2,3,4,5,6,7,
                                                  8,9,10,11,12,13,14,15);
    #pragma unroll
    for (int t = 0; t < 4; ++t) {
        // B fragment: col n = lane%16, 16 contiguous K (hs selects K half)
        v16h b = *(const v16h*)(bl + t * 16 * 32);
        acc[0][t] = __builtin_amdgcn_wmma_f32_16x16x32_f16(
            false, a0, false, b, (short)0, acc[0][t], false, false);
        acc[1][t] = __builtin_amdgcn_wmma_f32_16x16x32_f16(
            false, a1, false, b, (short)0, acc[1][t], false, false);
    }
}

// ---------------------------------------------------------------------------
// WMMA GEMM core: C = A(MxK) * B(NxK)^T, f16 inputs.
// Block tile 256(M) x 64(N); 8 waves; wave w owns rows [m0+32w, m0+32w+32).
// B tile (64n x 32k f16, 4KB) double-buffered in LDS, filled by async DMA.
// K loop manually unrolled x2 so the buffer parity is structural (one acc web).
// kTiles must be even.
// ---------------------------------------------------------------------------
__device__ __forceinline__ void gemm_core(const _Float16* __restrict__ A, int lda,
                                          const _Float16* __restrict__ B, int ldb,
                                          int kTiles, _Float16* lds,
                                          v8f acc[2][4], int aRow0, int n0) {
    const int tid  = threadIdx.x;
    const int lane = tid & 31;
    const int hs   = lane >> 4;          // half-wave select (A/B K-layout)
    const int lrow = tid >> 2;           // 0..63  (n within tile)
    const int lcol = (tid & 3) * 8;      // 0,8,16,24 (k chunk)

    const _Float16* bSrc = B + (size_t)(n0 + lrow) * ldb + lcol;
    const unsigned  ldsOff0 = (unsigned)(uintptr_t)(lds + lrow * 32 + lcol);
    const _Float16* aSrc0 = A + (size_t)aRow0 * lda + hs * 16;
    const _Float16* aSrc1 = aSrc0 + (size_t)16 * lda;
    const _Float16* bl0   = lds + (lane & 15) * 32 + hs * 16;
    const _Float16* bl1   = bl0 + 64 * 32;

    // prologue: DMA first B tile into buffer 0
    async_copy_b128(ldsOff0, bSrc);
    bSrc += 32;
    wait_asynccnt0();
    __syncthreads();

    #pragma unroll 1
    for (int kt = 0; kt < kTiles; kt += 2) {
        // DMA tile kt+1 into buffer 1, compute tile kt from buffer 0
        async_copy_b128(ldsOff0 + 4096u, bSrc);
        bSrc += 32;
        mma_step(aSrc0, aSrc1, bl0, acc);
        aSrc0 += 32;
        aSrc1 += 32;
        wait_asynccnt0();
        __syncthreads();

        // DMA tile kt+2 into buffer 0, compute tile kt+1 from buffer 1
        if (kt + 2 < kTiles) {
            async_copy_b128(ldsOff0, bSrc);
            bSrc += 32;
        }
        mma_step(aSrc0, aSrc1, bl1, acc);
        aSrc0 += 32;
        aSrc1 += 32;
        wait_asynccnt0();
        __syncthreads();
    }
}

// ---------------------------------------------------------------------------
// QKV projection: z = 0/1/2 -> Q/K/V.  Q,K row-major f16; V stored transposed
// per batch: Vt[b][d][s]  (so PV GEMM B-fragments are contiguous).
// ---------------------------------------------------------------------------
__global__ void __launch_bounds__(256) gemm_qkv(const _Float16* __restrict__ xh,
                                                const _Float16* __restrict__ Wh,
                                                _Float16* __restrict__ Qh,
                                                _Float16* __restrict__ Kh,
                                                _Float16* __restrict__ Vt) {
    __shared__ __align__(64) _Float16 lbs[2 * 64 * 32];
    const int z    = blockIdx.z;
    const _Float16* W = Wh + (size_t)z * DQ * DQ;
    const int wave = threadIdx.x >> 5;
    const int lane = threadIdx.x & 31;
    const int hs   = lane >> 4;
    const int n0   = blockIdx.x * 64;
    const int m0   = blockIdx.y * 256 + wave * 32;

    v8f acc[2][4] = {};
    gemm_core(xh, DQ, W, DQ, DQ / 32, lbs, acc, m0 + (lane & 15), n0);

    const int col0 = n0 + (lane & 15);
    if (z < 2) {
        _Float16* dst = (z == 0) ? Qh : Kh;
        #pragma unroll
        for (int mi = 0; mi < 2; ++mi) {
            const int rowB = m0 + mi * 16 + hs * 8;
            #pragma unroll
            for (int t = 0; t < 4; ++t)
                #pragma unroll
                for (int r = 0; r < 8; ++r)
                    dst[(size_t)(rowB + r) * DQ + col0 + t * 16] =
                        (_Float16)acc[mi][t][r];
        }
    } else {
        #pragma unroll
        for (int mi = 0; mi < 2; ++mi) {
            const int rowB = m0 + mi * 16 + hs * 8;
            #pragma unroll
            for (int t = 0; t < 4; ++t)
                #pragma unroll
                for (int r = 0; r < 8; ++r) {
                    int mm = rowB + r;
                    int b  = mm >> 11;          // / SQ
                    int s  = mm & (SQ - 1);
                    Vt[((size_t)b * DQ + (col0 + t * 16)) * SQ + s] =
                        (_Float16)acc[mi][t][r];
                }
        }
    }
}

// ---------------------------------------------------------------------------
// Generic batched f16 GEMM with f32 (scaled) output: scores and PV.
// ---------------------------------------------------------------------------
__global__ void __launch_bounds__(256) gemm_f32(const _Float16* __restrict__ Ab, long sAz, int lda,
                                                const _Float16* __restrict__ Bb, long sBz, int ldb,
                                                float* __restrict__ Cb, long sCz, int ldc,
                                                int kTiles, float scale) {
    __shared__ __align__(64) _Float16 lbs[2 * 64 * 32];
    const _Float16* A  = Ab + (size_t)blockIdx.z * sAz;
    const _Float16* Bm = Bb + (size_t)blockIdx.z * sBz;
    float* C = Cb + (size_t)blockIdx.z * sCz;
    const int wave = threadIdx.x >> 5;
    const int lane = threadIdx.x & 31;
    const int hs   = lane >> 4;
    const int n0   = blockIdx.x * 64;
    const int m0   = blockIdx.y * 256 + wave * 32;

    v8f acc[2][4] = {};
    gemm_core(A, lda, Bm, ldb, kTiles, lbs, acc, m0 + (lane & 15), n0);

    const int col0 = n0 + (lane & 15);
    #pragma unroll
    for (int mi = 0; mi < 2; ++mi) {
        const int rowB = m0 + mi * 16 + hs * 8;
        #pragma unroll
        for (int t = 0; t < 4; ++t)
            #pragma unroll
            for (int r = 0; r < 8; ++r)
                C[(size_t)(rowB + r) * ldc + col0 + t * 16] = acc[mi][t][r] * scale;
    }
}

// ---------------------------------------------------------------------------
// Row softmax over 2048 f32, in place; also emit f16 copy of probabilities.
// ---------------------------------------------------------------------------
__global__ void __launch_bounds__(256) softmax_rows(float* __restrict__ Wt,
                                                    _Float16* __restrict__ Ph) {
    __shared__ float red[8];
    const int row = blockIdx.x;
    const int b   = blockIdx.y;
    const size_t base = ((size_t)b * SQ + row) * SQ;
    float4* p4 = (float4*)(Wt + base);
    const int tid = threadIdx.x;

    float4 v0 = p4[tid * 2];
    float4 v1 = p4[tid * 2 + 1];

    float m = fmaxf(fmaxf(fmaxf(v0.x, v0.y), fmaxf(v0.z, v0.w)),
                    fmaxf(fmaxf(v1.x, v1.y), fmaxf(v1.z, v1.w)));
    #pragma unroll
    for (int o = 16; o > 0; o >>= 1) m = fmaxf(m, __shfl_xor(m, o));
    if ((tid & 31) == 0) red[tid >> 5] = m;
    __syncthreads();
    m = red[0];
    #pragma unroll
    for (int i = 1; i < 8; ++i) m = fmaxf(m, red[i]);
    __syncthreads();

    float e[8];
    e[0] = __expf(v0.x - m); e[1] = __expf(v0.y - m);
    e[2] = __expf(v0.z - m); e[3] = __expf(v0.w - m);
    e[4] = __expf(v1.x - m); e[5] = __expf(v1.y - m);
    e[6] = __expf(v1.z - m); e[7] = __expf(v1.w - m);
    float s = ((e[0] + e[1]) + (e[2] + e[3])) + ((e[4] + e[5]) + (e[6] + e[7]));
    #pragma unroll
    for (int o = 16; o > 0; o >>= 1) s += __shfl_xor(s, o);
    if ((tid & 31) == 0) red[tid >> 5] = s;
    __syncthreads();
    s = ((red[0] + red[1]) + (red[2] + red[3])) + ((red[4] + red[5]) + (red[6] + red[7]));
    const float inv = 1.0f / s;

    float4 o0 = make_float4(e[0] * inv, e[1] * inv, e[2] * inv, e[3] * inv);
    float4 o1 = make_float4(e[4] * inv, e[5] * inv, e[6] * inv, e[7] * inv);
    p4[tid * 2]     = o0;
    p4[tid * 2 + 1] = o1;
    v8h h = { (_Float16)o0.x, (_Float16)o0.y, (_Float16)o0.z, (_Float16)o0.w,
              (_Float16)o1.x, (_Float16)o1.y, (_Float16)o1.z, (_Float16)o1.w };
    ((v8h*)(Ph + base))[tid] = h;
}

// ---------------------------------------------------------------------------
extern "C" void kernel_launch(void* const* d_in, const int* in_sizes, int n_in,
                              void* d_out, int out_size, void* d_ws, size_t ws_size,
                              hipStream_t stream) {
    (void)in_sizes; (void)n_in; (void)out_size; (void)ws_size;
    const float* x  = (const float*)d_in[0];
    const float* Wq = (const float*)d_in[1];
    const float* Wk = (const float*)d_in[2];
    const float* Wv = (const float*)d_in[3];

    float* outV = (float*)d_out;                        // [B,S,D] weighted values
    float* outW = outV + (size_t)BQ * SQ * DQ;          // [B,S,S] attention weights

    char* ws = (char*)d_ws;
    size_t off = 0;
    auto wsAlloc = [&](size_t bytes) -> char* {
        char* p = ws + off;
        off += (bytes + 255) & ~(size_t)255;
        return p;
    };
    _Float16* xh = (_Float16*)wsAlloc((size_t)BQ * SQ * DQ * 2);
    _Float16* Wh = (_Float16*)wsAlloc((size_t)3 * DQ * DQ * 2);
    _Float16* Qh = (_Float16*)wsAlloc((size_t)BQ * SQ * DQ * 2);
    _Float16* Kh = (_Float16*)wsAlloc((size_t)BQ * SQ * DQ * 2);
    _Float16* Vt = (_Float16*)wsAlloc((size_t)BQ * SQ * DQ * 2);
    _Float16* Ph = (_Float16*)wsAlloc((size_t)BQ * SQ * SQ * 2);

    // 1) convert inputs to f16
    {
        int n4 = (BQ * SQ * DQ) / 4;
        cvt_f32_to_f16<<<(n4 + 255) / 256, 256, 0, stream>>>(x, xh, n4);
        int w4 = (DQ * DQ) / 4;
        cvt_f32_to_f16<<<(w4 + 255) / 256, 256, 0, stream>>>(Wq, Wh, w4);
        cvt_f32_to_f16<<<(w4 + 255) / 256, 256, 0, stream>>>(Wk, Wh + (size_t)DQ * DQ, w4);
        cvt_f32_to_f16<<<(w4 + 255) / 256, 256, 0, stream>>>(Wv, Wh + (size_t)2 * DQ * DQ, w4);
    }

    // 2) Q/K/V projections (M=B*S=16384, N=768, K=768), z selects weight
    gemm_qkv<<<dim3(DQ / 64, (BQ * SQ) / 256, 3), 256, 0, stream>>>(xh, Wh, Qh, Kh, Vt);

    // 3) scores = (Q K^T) / sqrt(D) per batch -> f32 into weights output region
    const float scale = 0.03608439182435161f;  // 1/sqrt(768)
    gemm_f32<<<dim3(SQ / 64, SQ / 256, BQ), 256, 0, stream>>>(
        Qh, (long)SQ * DQ, DQ,
        Kh, (long)SQ * DQ, DQ,
        outW, (long)SQ * SQ, SQ,
        DQ / 32, scale);

    // 4) row softmax in place + f16 copy for PV GEMM
    softmax_rows<<<dim3(SQ, BQ), 256, 0, stream>>>(outW, Ph);

    // 5) weighted values = P V  (M=2048, N=768, K=2048), B = Vt (d-major)
    gemm_f32<<<dim3(DQ / 64, SQ / 256, BQ), 256, 0, stream>>>(
        Ph, (long)SQ * SQ, SQ,
        Vt, (long)DQ * SQ, SQ,
        outV, (long)SQ * DQ, DQ,
        SQ / 32, 1.0f);
}
